// GCNModelWPathways_89541478187036
// MI455X (gfx1250) — matrix operand order repaired and checked by reading, not compiled, not run on hardware
//
#include <hip/hip_runtime.h>
#include <hip/hip_bf16.h>
#include <math.h>

typedef __attribute__((ext_vector_type(16))) _Float16 v16h;
typedef __attribute__((ext_vector_type(8)))  float    v8f;

#define N_GENES 15135
#define N_CMT   1000
#define HIDF    64
#define BS      8

// ---------------------------------------------------------------------------
// degree accumulation (self-loop added analytically later): deg[dst[e]] += 1
__global__ void deg_accum_kernel(const int* __restrict__ dst, int E,
                                 float* __restrict__ deg) {
    int e = blockIdx.x * blockDim.x + threadIdx.x;
    if (e < E) atomicAdd(&deg[dst[e]], 1.0f);
}

// deg -> dinv in place: dinv = 1/sqrt(deg + 1)   (self loop guarantees deg>=1)
__global__ void dinv_kernel(float* __restrict__ deg, int n) {
    int i = blockIdx.x * blockDim.x + threadIdx.x;
    if (i < n) deg[i] = rsqrtf(deg[i] + 1.0f);
}

// norm[e] = dinv[src]*dinv[dst]
__global__ void norm_kernel(const int* __restrict__ src, const int* __restrict__ dst,
                            const float* __restrict__ dinv, float* __restrict__ nrm, int E) {
    int e = blockIdx.x * blockDim.x + threadIdx.x;
    if (e < E) nrm[e] = dinv[src[e]] * dinv[dst[e]];
}

// ---------------------------------------------------------------------------
// Convert the three 64x64 weights f32->f16 AND pre-swizzle into per-lane WMMA
// B-fragment order:  frag[layer][nt][kh][lane][j]  (j = 0..15 contiguous), so
// the GEMM loads each B fragment as a single contiguous 32-byte v16h.
// Element (nt,kh,lane,j) holds W[k][col], k = kh*32 + (j>>3)*16 + (lane>>4)*8 + (j&7),
// col = nt*16 + (lane&15)   (dense 16-bit B 32x16 layout, ISA 7.12.2).
__global__ void wfrag_kernel(const float* __restrict__ w1, const float* __restrict__ w2,
                             const float* __restrict__ w3, _Float16* __restrict__ o) {
    int i = blockIdx.x * blockDim.x + threadIdx.x;
    if (i >= 3 * 4096) return;
    int layer = i >> 12;
    int t = i & 4095;               // t = ((nt*2+kh)*32 + lane)*16 + j
    int j    = t & 15;
    int lane = (t >> 4) & 31;
    int kh   = (t >> 9) & 1;
    int nt   = t >> 10;
    int k   = kh * 32 + ((j >> 3) << 4) + ((lane >> 4) << 3) + (j & 7);
    int col = nt * 16 + (lane & 15);
    const float* w = (layer == 0) ? w1 : (layer == 1) ? w2 : w3;
    o[i] = (_Float16)w[k * 64 + col];
}

// ---------------------------------------------------------------------------
// WMMA GEMM: H = X(f32->f16) @ W(f16), f32 accumulate.
// Epilogue: Hbuf = H ; Obuf = H*dinv[n]^2 + bias  (self-loop + bias init)
// Grid: (tiles-per-batch, batch). One wave = 16 rows x 64 cols = 8 wmma ops.
// Batch offset folded into scalar base pointers; per-row stores use immediate
// offsets; only the single partial tile per batch takes the predicated path.
__global__ __launch_bounds__(256)
void gemm_wmma_kernel(const float* __restrict__ X, const _Float16* __restrict__ Wfrag,
                      const float* __restrict__ bias, const float* __restrict__ dinv,
                      float* __restrict__ Hbuf, float* __restrict__ Obuf) {
    const int lane = threadIdx.x & 31;
    const int lo   = lane & 15;
    const int hi   = lane >> 4;
    const int tileBase = blockIdx.x * 128 + (threadIdx.x >> 5) * 16; // row in batch

    const size_t bOff64 = (size_t)blockIdx.y * ((size_t)N_GENES * 64); // uniform
    const float* Xb = X    + bOff64;
    float*       Hp = Hbuf + bOff64;
    float*       Op = Obuf + bOff64;

    // ---- A fragments (16-bit A 16x32 layout). Out-of-range rows clamped:
    // WMMA output rows are independent, and clamped rows are never stored.
    int nA = tileBase + lo;
    nA = (nA < N_GENES) ? nA : (N_GENES - 1);
    const float* xr = Xb + nA * 64;
    v16h a0, a1;
#pragma unroll
    for (int v = 0; v < 8; ++v) {
        int k0 = ((v >> 2) << 4) + (hi << 3) + ((v & 3) << 1);
        float2 f0 = *(const float2*)(xr + k0);
        float2 f1 = *(const float2*)(xr + 32 + k0);
        a0[2 * v]     = (_Float16)f0.x;
        a0[2 * v + 1] = (_Float16)f0.y;
        a1[2 * v]     = (_Float16)f1.x;
        a1[2 * v + 1] = (_Float16)f1.y;
    }

    // ---- per-lane self-loop norms (clamped loads, no predication)
    const int rBase = tileBase + (hi << 3);
    float sn[8];
#pragma unroll
    for (int r = 0; r < 8; ++r) {
        int nc = rBase + r;
        nc = (nc < N_GENES) ? nc : (N_GENES - 1);
        float dv = dinv[nc];
        sn[r] = dv * dv;
    }

    const bool fullTile = (tileBase + 16) <= N_GENES;   // wave-uniform

#pragma unroll
    for (int nt = 0; nt < 4; ++nt) {
        // B fragments: one contiguous 32B load each (L2-resident broadcast)
        v16h b0 = *(const v16h*)(Wfrag + ((nt * 2 + 0) * 32 + lane) * 16);
        v16h b1 = *(const v16h*)(Wfrag + ((nt * 2 + 1) * 32 + lane) * 16);
        v8f acc = {};
        acc = __builtin_amdgcn_wmma_f32_16x16x32_f16(false, a0, false, b0,
                                                     (short)0, acc, false, false);
        acc = __builtin_amdgcn_wmma_f32_16x16x32_f16(false, a1, false, b1,
                                                     (short)0, acc, false, false);
        const int col  = nt * 16 + lo;
        const float bc = bias[col];
        const int vo   = rBase * 64 + col;              // 32-bit lane offset
        if (fullTile) {
#pragma unroll
            for (int r = 0; r < 8; ++r) {               // branch-free hot path
                Hp[vo + r * 64] = acc[r];
                Op[vo + r * 64] = fmaf(acc[r], sn[r], bc);
            }
        } else {
#pragma unroll
            for (int r = 0; r < 8; ++r) {               // last tile per batch only
                if (rBase + r < N_GENES) {
                    Hp[vo + r * 64] = acc[r];
                    Op[vo + r * 64] = fmaf(acc[r], sn[r], bc);
                }
            }
        }
    }
}

// ---------------------------------------------------------------------------
// Edge scatter: Obuf[b,dst,h] += Hbuf[b,src,h]*norm[e]; thread = (edge, 4 feats)
__global__ void edge_scatter_kernel(const int* __restrict__ src, const int* __restrict__ dst,
                                    const float* __restrict__ nrm,
                                    const float* __restrict__ Hbuf, float* __restrict__ Obuf,
                                    int E) {
    long idx = (long)blockIdx.x * blockDim.x + threadIdx.x;
    int e  = (int)(idx >> 4);
    int h4 = (int)(idx & 15) << 2;
    if (e >= E) return;
    int s = src[e], d = dst[e];
    float w = nrm[e];
    const float* hp = Hbuf + (long)s * 64 + h4;
    float* op = Obuf + (long)d * 64 + h4;
    const long stride = (long)N_GENES * 64;
#pragma unroll
    for (int b = 0; b < BS; ++b) {
        float4 v = *(const float4*)(hp + b * stride);
        float* o = op + b * stride;
        atomicAdd(o + 0, v.x * w);
        atomicAdd(o + 1, v.y * w);
        atomicAdd(o + 2, v.z * w);
        atomicAdd(o + 3, v.w * w);
    }
}

__global__ void relu4_kernel(float4* __restrict__ x, long n4) {
    long i = (long)blockIdx.x * blockDim.x + threadIdx.x;
    if (i < n4) {
        float4 v = x[i];
        v.x = fmaxf(v.x, 0.f); v.y = fmaxf(v.y, 0.f);
        v.z = fmaxf(v.z, 0.f); v.w = fmaxf(v.w, 0.f);
        x[i] = v;
    }
}

// ---------------------------------------------------------------------------
// Pathway pooling fused with fc(192->1)  (fc is linear, so it commutes with
// the scatter-mean; mean division deferred to fc1):
//   hcmt[b,c] += sum_h ( x1*fcW[3h] + x2*fcW[3h+1] + x3*fcW[3h+2] ) at row[p]
__global__ void pool_fc_kernel(const int* __restrict__ prow, const int* __restrict__ pcol,
                               const float* __restrict__ x1, const float* __restrict__ x2,
                               const float* __restrict__ x3, const float* __restrict__ fcW,
                               float* __restrict__ hcmt, float* __restrict__ cnt, int P) {
    long idx = (long)blockIdx.x * blockDim.x + threadIdx.x;
    int p = (int)(idx >> 3);
    int b = (int)(idx & 7);
    if (p >= P) return;
    int r = prow[p], c = pcol[p];
    long base = ((long)b * N_GENES + r) * 64;
    const float4* p1 = (const float4*)(x1 + base);
    const float4* p2 = (const float4*)(x2 + base);
    const float4* p3 = (const float4*)(x3 + base);
    float d = 0.0f;
#pragma unroll
    for (int q = 0; q < 16; ++q) {
        float4 v1 = p1[q], v2 = p2[q], v3 = p3[q];
        const float* fw = fcW + 12 * q;
        d += v1.x * fw[0]  + v2.x * fw[1]  + v3.x * fw[2]
           + v1.y * fw[3]  + v2.y * fw[4]  + v3.y * fw[5]
           + v1.z * fw[6]  + v2.z * fw[7]  + v3.z * fw[8]
           + v1.w * fw[9]  + v2.w * fw[10] + v3.w * fw[11];
    }
    atomicAdd(&hcmt[b * N_CMT + c], d);
    if (b == 0) atomicAdd(&cnt[c], 1.0f);
}

// h1[b,j] = relu( sum_c (hcmt[b,c]/max(cnt,1) + fcb) * l1W[c,j] + l1b[j] )
__global__ void fc1_kernel(const float* __restrict__ hcmt, const float* __restrict__ cnt,
                           const float* __restrict__ fcb, const float* __restrict__ l1W,
                           const float* __restrict__ l1b, float* __restrict__ h1) {
    int idx = blockIdx.x * blockDim.x + threadIdx.x;
    if (idx >= BS * 128) return;
    int b = idx >> 7, j = idx & 127;
    float fb = fcb[0];
    float s = 0.0f;
    for (int c = 0; c < N_CMT; ++c) {
        float hv = hcmt[b * N_CMT + c] / fmaxf(cnt[c], 1.0f) + fb;
        s += hv * l1W[c * 128 + j];
    }
    h1[b * 128 + j] = fmaxf(s + l1b[j], 0.0f);
}

// logits + log_softmax (2 classes)
__global__ void fc2_kernel(const float* __restrict__ h1, const float* __restrict__ l2W,
                           const float* __restrict__ l2b, float* __restrict__ out) {
    int b = threadIdx.x;
    if (b >= BS) return;
    float z0 = l2b[0], z1 = l2b[1];
    for (int j = 0; j < 128; ++j) {
        float v = h1[b * 128 + j];
        z0 += v * l2W[j * 2];
        z1 += v * l2W[j * 2 + 1];
    }
    float m = fmaxf(z0, z1);
    float lse = m + logf(expf(z0 - m) + expf(z1 - m));
    out[b * 2]     = z0 - lse;
    out[b * 2 + 1] = z1 - lse;
}

// ---------------------------------------------------------------------------
extern "C" void kernel_launch(void* const* d_in, const int* in_sizes, int n_in,
                              void* d_out, int out_size, void* d_ws, size_t ws_size,
                              hipStream_t stream) {
    const float* x    = (const float*)d_in[0];
    const int*   edge = (const int*)d_in[2];
    const int*   prow = (const int*)d_in[3];
    const int*   pcol = (const int*)d_in[4];
    const float* W1 = (const float*)d_in[5];  const float* b1 = (const float*)d_in[6];
    const float* W2 = (const float*)d_in[7];  const float* b2 = (const float*)d_in[8];
    const float* W3 = (const float*)d_in[9];  const float* b3 = (const float*)d_in[10];
    const float* fcW = (const float*)d_in[11]; const float* fcb = (const float*)d_in[12];
    const float* l1W = (const float*)d_in[13]; const float* l1b = (const float*)d_in[14];
    const float* l2W = (const float*)d_in[15]; const float* l2b = (const float*)d_in[16];

    const int E = in_sizes[2] / 2;
    const int P = in_sizes[3];
    const long M = (long)BS * N_GENES;
    const int* esrc = edge;
    const int* edst = edge + E;

    // bump-allocate workspace (256B aligned)
    char* wsp = (char*)d_ws;
    auto alloc = [&](size_t bytes) -> char* {
        char* p = wsp;
        wsp += (bytes + 255) & ~(size_t)255;
        return p;
    };
    const size_t actB = (size_t)M * 64 * sizeof(float);   // ~31 MB
    float*    X1   = (float*)alloc(actB);
    float*    X2   = (float*)alloc(actB);
    float*    X3   = (float*)alloc(actB);
    float*    Hb   = (float*)alloc(actB);
    float*    nrm  = (float*)alloc((size_t)E * sizeof(float));
    float*    dinv = (float*)alloc((size_t)N_GENES * sizeof(float)); // deg -> dinv in place
    _Float16* Wh   = (_Float16*)alloc(3 * 4096 * sizeof(_Float16));
    float*    hcmt = (float*)alloc((size_t)BS * N_CMT * sizeof(float));
    float*    cnt  = (float*)alloc((size_t)N_CMT * sizeof(float));
    float*    h1   = (float*)alloc((size_t)BS * 128 * sizeof(float));

    // zero accumulation targets (graph-capture-legal stream memsets)
    hipMemsetAsync(dinv, 0, (size_t)N_GENES * sizeof(float), stream);
    hipMemsetAsync(hcmt, 0, (size_t)BS * N_CMT * sizeof(float), stream);
    hipMemsetAsync(cnt,  0, (size_t)N_CMT * sizeof(float), stream);

    // graph normalization + weight fragment pre-swizzle
    deg_accum_kernel<<<(E + 255) / 256, 256, 0, stream>>>(edst, E, dinv);
    dinv_kernel<<<(N_GENES + 255) / 256, 256, 0, stream>>>(dinv, N_GENES);
    norm_kernel<<<(E + 255) / 256, 256, 0, stream>>>(esrc, edst, dinv, nrm, E);
    wfrag_kernel<<<(3 * 4096 + 255) / 256, 256, 0, stream>>>(W1, W2, W3, Wh);

    const dim3 gemmGrid((N_GENES + 127) / 128, BS);       // 119 x 8
    const int scatBlocks = (int)(((long)E * 16 + 255) / 256);
    const long n4 = M * 16;                               // M*64/4 float4s
    const int reluBlocks = (int)((n4 + 255) / 256);

    // layer 1
    gemm_wmma_kernel<<<gemmGrid, 256, 0, stream>>>(x, Wh, b1, dinv, Hb, X1);
    edge_scatter_kernel<<<scatBlocks, 256, 0, stream>>>(esrc, edst, nrm, Hb, X1, E);
    relu4_kernel<<<reluBlocks, 256, 0, stream>>>((float4*)X1, n4);
    // layer 2
    gemm_wmma_kernel<<<gemmGrid, 256, 0, stream>>>(X1, Wh + 4096, b2, dinv, Hb, X2);
    edge_scatter_kernel<<<scatBlocks, 256, 0, stream>>>(esrc, edst, nrm, Hb, X2, E);
    relu4_kernel<<<reluBlocks, 256, 0, stream>>>((float4*)X2, n4);
    // layer 3
    gemm_wmma_kernel<<<gemmGrid, 256, 0, stream>>>(X2, Wh + 8192, b3, dinv, Hb, X3);
    edge_scatter_kernel<<<scatBlocks, 256, 0, stream>>>(esrc, edst, nrm, Hb, X3, E);
    relu4_kernel<<<reluBlocks, 256, 0, stream>>>((float4*)X3, n4);

    // pathway pooling fused with fc(192->1), then head MLP + log_softmax
    pool_fc_kernel<<<(int)(((long)P * 8 + 255) / 256), 256, 0, stream>>>(
        prow, pcol, X1, X2, X3, fcW, hcmt, cnt, P);
    fc1_kernel<<<(BS * 128 + 255) / 256, 256, 0, stream>>>(hcmt, cnt, fcb, l1W, l1b, h1);
    fc2_kernel<<<1, 32, 0, stream>>>(h1, l2W, l2b, (float*)d_out);
}
